// RFWPlayer_30889404792853
// MI455X (gfx1250) — compile-verified, matrix operand order checked by reading, not run on hardware
//
#include <hip/hip_runtime.h>
#include <hip/hip_bf16.h>

// ---------- CDNA5 WMMA vector types ----------
typedef __attribute__((ext_vector_type(2)))  float    v2f;
typedef __attribute__((ext_vector_type(8)))  float    v8f;
typedef __attribute__((ext_vector_type(16))) _Float16 v16h;

// ---------- problem constants ----------
#define SLEN 512
#define BSZ  16
#define DIM  1024
#define NH   16
#define DH   64
#define NQ   3088          // H*(3*d+1) = 16*193
#define NQP  3200          // padded to 200 16-wide tiles (zero rows in Bt)
#define ROWS 8192          // SLEN*BSZ
#define HW   193           // per-head qkvb width

// workspace offsets (bytes, 256-aligned)
#define OFF_HN     ((size_t)0)                      // f16 [8192][1024]   16,777,216
#define OFF_WSLOWT ((size_t)16777216)               // f16 [3200][1024]    6,553,600  (W_slow^T, padded)
#define OFF_WOUTT  ((size_t)23330816)               // f16 [1024][1024]    2,097,152  (W_out^T)
#define OFF_QKVB   ((size_t)25427968)               // f32 [8192][3088]  101,187,584
#define OFF_OUTS   ((size_t)126615552)              // f16 [8192][1024]   16,777,216

// =====================================================================
// Tiled transpose + f32->f16: out[c][r] = (f16) in[r][c], zero-padding
// output rows in [cols, out_rows). block = 256 (32x8), tile 32x33.
// =====================================================================
__global__ __launch_bounds__(256) void transpose_f16_kernel(const float* __restrict__ in,
                                                            _Float16* __restrict__ out,
                                                            int rows, int cols, int out_rows) {
    __shared__ float tile[32][33];
    const int c0 = blockIdx.x * 32, r0 = blockIdx.y * 32;
    const int tx = threadIdx.x & 31, ty = threadIdx.x >> 5;
#pragma unroll
    for (int i = ty; i < 32; i += 8) {
        const int r = r0 + i, c = c0 + tx;
        tile[i][tx] = (c < cols) ? in[(size_t)r * cols + c] : 0.f;
    }
    __syncthreads();
#pragma unroll
    for (int i = ty; i < 32; i += 8) {
        const int c = c0 + i;                     // output row (= input col, maybe pad)
        if (c < out_rows) out[(size_t)c * rows + r0 + tx] = (_Float16)tile[tx][i];
    }
}

// =====================================================================
// LayerNorm(x) -> f16, one row (1024) per block of 256 threads
// =====================================================================
__global__ __launch_bounds__(256) void ln_kernel(const float* __restrict__ x,
                                                 const float* __restrict__ g,
                                                 const float* __restrict__ b,
                                                 _Float16* __restrict__ hn) {
    const int row  = blockIdx.x;
    const int lane = threadIdx.x & 31, wv = threadIdx.x >> 5;
    const float* xr = x + (size_t)row * DIM;
    float v[4], s = 0.f, ss = 0.f;
#pragma unroll
    for (int i = 0; i < 4; ++i) {
        v[i] = xr[threadIdx.x + 256 * i];
        s += v[i]; ss += v[i] * v[i];
    }
#pragma unroll
    for (int off = 16; off; off >>= 1) {        // wave32 reduction
        s  += __shfl_xor(s,  off, 32);
        ss += __shfl_xor(ss, off, 32);
    }
    __shared__ float ps[8], pss[8];
    __shared__ float mu_s, rs_s;
    if (lane == 0) { ps[wv] = s; pss[wv] = ss; }
    __syncthreads();
    if (threadIdx.x == 0) {
        float S = 0.f, SS = 0.f;
        for (int i = 0; i < 8; ++i) { S += ps[i]; SS += pss[i]; }
        float mu  = S * (1.f / DIM);
        float var = SS * (1.f / DIM) - mu * mu;
        mu_s = mu; rs_s = rsqrtf(var + 1e-5f);
    }
    __syncthreads();
    const float mu = mu_s, rs = rs_s;
#pragma unroll
    for (int i = 0; i < 4; ++i) {
        int c = threadIdx.x + 256 * i;
        hn[(size_t)row * DIM + c] = (_Float16)((v[i] - mu) * rs * g[c] + b[c]);
    }
}

// =====================================================================
// GEMM1: qkvb[8192][3088] = hn_f16[8192][1024] @ Wslow[1024][3088]
// Bt is W_slow^T padded to 3200 zero rows -> branch-free K loop.
// wave: 16(M) x 8x16(N); inner loop = groups of 4: clause of 8 b128
// loads, then 4 back-to-back wmmas (load/compute overlap across groups).
// =====================================================================
__global__ __launch_bounds__(128) void gemm_qkvb_kernel(const _Float16* __restrict__ A,
                                                        const _Float16* __restrict__ Bt,
                                                        float* __restrict__ C) {
    const int lane = threadIdx.x & 31, wave = threadIdx.x >> 5;
    const int lm = lane & 15, hi = lane >> 4, koff = hi * 8;
    const int m0  = blockIdx.y * 16;
    const int nt0 = (blockIdx.x * 4 + wave) * 8;
    if (nt0 >= 200) return;                       // fully-dead wave column (uniform)
    v8f acc[8] = {};
    const _Float16* aprow = A + (size_t)(m0 + lm) * DIM + koff;
    const _Float16* bbase = Bt + (size_t)(nt0 * 16 + lm) * DIM + koff;
    for (int kk = 0; kk < DIM; kk += 32) {
        v16h a;
        const _Float16* ap = aprow + kk;
#pragma unroll
        for (int i = 0; i < 8; ++i) { a[i] = ap[i]; a[8 + i] = ap[16 + i]; }
#pragma unroll
        for (int jj = 0; jj < 8; jj += 4) {
            v16h bf[4];
#pragma unroll
            for (int j = 0; j < 4; ++j) {
                const _Float16* bp = bbase + (size_t)(jj + j) * 16 * DIM + kk;
#pragma unroll
                for (int i = 0; i < 8; ++i) { bf[j][i] = bp[i]; bf[j][8 + i] = bp[16 + i]; }
            }
#pragma unroll
            for (int j = 0; j < 4; ++j)
                acc[jj + j] = __builtin_amdgcn_wmma_f32_16x16x32_f16(
                    false, a, false, bf[j], (short)0, acc[jj + j], false, false);
        }
    }
#pragma unroll
    for (int j = 0; j < 8; ++j) {
        const int nt = nt0 + j;
        if (nt >= 193) continue;                  // store guard only (once)
        const int col = nt * 16 + lm;
#pragma unroll
        for (int r = 0; r < 8; ++r) {             // C/D layout: VGPR r -> M = r + 8*hi
            const int m = r + hi * 8;
            C[(size_t)(m0 + m) * NQ + col] = acc[j][r];
        }
    }
}

// =====================================================================
// Recurrent fast-weight scan.
// grid = 32 blocks: block handles (head h, batch half b0..b0+7); 16 waves.
// LDS: Rq/Rk/Rv (48KB, async-staged via ASYNCcnt) + Wf[8][64][65] (133KB).
// Phase A: Hprev(16x64)@R via v_wmma_f32_16x16x4_f32 (rows 8..15 zero-padded).
// Phase B: per-batch delta-rule update, 2 waves/batch, 1 row/lane, LDS stride
//          65 => bank = (row+e)%64, conflict-free; kr/qr reads are broadcasts.
// =====================================================================
#define SM_R     0            // 3*4096 floats
#define SM_RB    12288        // 64
#define SM_H     12352        // 16 rows * 66 stride
#define SM_Q     13408        // 16*64
#define SM_K     14432
#define SM_V     15456
#define SM_BETA  16480        // 8 (+pad)
#define SM_WF    16512        // 8 * 64 * 65
#define SM_TOTAL 49792        // floats -> 199,168 bytes

__global__ __launch_bounds__(512) void scan_kernel(const float* __restrict__ qkvb,
                                                   const float* __restrict__ Rq,
                                                   const float* __restrict__ Rk,
                                                   const float* __restrict__ Rv,
                                                   const float* __restrict__ rb,
                                                   const float* __restrict__ state0,
                                                   _Float16* __restrict__ outs) {
    extern __shared__ float sm[];
    float* sR    = sm + SM_R;
    float* srb   = sm + SM_RB;
    float* sh    = sm + SM_H;
    float* sq    = sm + SM_Q;
    float* sk    = sm + SM_K;
    float* sv    = sm + SM_V;
    float* sbeta = sm + SM_BETA;
    float* sW    = sm + SM_WF;

    const int h  = blockIdx.x >> 1;
    const int b0 = (blockIdx.x & 1) * 8;
    const int tid = threadIdx.x, lane = tid & 31, wv = tid >> 5;

    // ---- init: async-stage per-head R matrices into LDS (16B per lane/iter),
    //      tracked by ASYNCcnt; then rb, zero Wf and Hprev ----
    for (int i = tid; i < 3072; i += 512) {            // 3 mats * 1024 chunks of 16B
        const int mat = i >> 10, off = (i & 1023) * 4; // float offset within matrix
        const float* src = ((mat == 0) ? Rq : (mat == 1) ? Rk : Rv) + (size_t)h * 4096 + off;
        unsigned dst = (unsigned)(uintptr_t)(sR + mat * 4096 + off);
        asm volatile("global_load_async_to_lds_b128 %0, %1, off"
                     :: "v"(dst), "v"(src) : "memory");
    }
    asm volatile("s_wait_asynccnt 0x0" ::: "memory");
    if (tid < 64) srb[tid] = rb[(size_t)h * 64 + tid];
    for (int i = tid; i < 16 * 66; i += 512) sh[i] = 0.f;
    for (int i = tid; i < 8 * 64 * 65; i += 512) sW[i] = 0.f;
    __syncthreads();
    {   // Hprev rows 0..7 <- state0[b0+b][h][0][:]   (rows 8..15 stay zero)
        const int b = tid >> 6, d = tid & 63;          // 512 threads = 8*64 exactly
        sh[b * 66 + d] = state0[((size_t)(b0 + b) * NH + h) * DH + d];
    }
    __syncthreads();

    for (int t = 0; t < SLEN; ++t) {
        // ---------------- Phase A: projections via f32 WMMA ----------------
        if (wv < 12) {
            const int mat = wv >> 2;                 // 0=q 1=k 2=v
            const int nt  = wv & 3;                  // 16-wide N tile of 64
            const int lm = lane & 15, kb = (lane >> 4) * 2;
            const float* Rm = sR + mat * 4096;
            v8f c = {};
#pragma unroll
            for (int k0 = 0; k0 < 64; k0 += 4) {
                v2f a, bfr;
                // A 16x4 f32 layout: lane m=lm; VGPR0/1 hold K=kb,kb+1
                a.x   = sh[lm * 66 + k0 + kb];
                a.y   = sh[lm * 66 + k0 + kb + 1];
                // B 4x16 f32 layout (mirror): lane n=lm; K=kb,kb+1
                bfr.x = Rm[(k0 + kb)     * 64 + nt * 16 + lm];
                bfr.y = Rm[(k0 + kb + 1) * 64 + nt * 16 + lm];
                c = __builtin_amdgcn_wmma_f32_16x16x4_f32(
                    false, a, false, bfr, (short)0, c, false, false);
            }
            if (lane < 16) {                         // lanes 0-15 hold M=0..7 (valid batches)
                float* dst = (mat == 0) ? sq : (mat == 1) ? sk : sv;
                const int e = nt * 16 + lm;
                const size_t qbase = ((size_t)(t * BSZ + b0)) * NQ + h * HW + mat * 64;
#pragma unroll
                for (int r = 0; r < 8; ++r)
                    dst[r * 64 + e] = c[r] + qkvb[qbase + (size_t)r * NQ + e];
            }
        } else if (wv == 12) {
            // beta gate: sigmoid(bt + <hprev, rb>) for 8 batches
            const int b = lane >> 2, part = lane & 3;
            float s = 0.f;
            for (int d = part * 16; d < part * 16 + 16; ++d) s += sh[b * 66 + d] * srb[d];
            s += __shfl_xor(s, 1, 32);
            s += __shfl_xor(s, 2, 32);
            if (part == 0) {
                float bt = qkvb[((size_t)(t * BSZ + b0 + b)) * NQ + h * HW + 192];
                sbeta[b] = 1.f / (1.f + expf(-(bt + s)));
            }
        }
        __syncthreads();

        // ---------------- softmax rows (q: waves 0-7, k: waves 8-15) -------
        {
            float* row = (wv < 8) ? (sq + wv * 64) : (sk + (wv - 8) * 64);
            float x0 = row[lane], x1 = row[lane + 32];
            float m = fmaxf(x0, x1);
#pragma unroll
            for (int off = 16; off; off >>= 1) m = fmaxf(m, __shfl_xor(m, off, 32));
            float e0 = expf(x0 - m), e1 = expf(x1 - m);
            float s = e0 + e1;
#pragma unroll
            for (int off = 16; off; off >>= 1) s += __shfl_xor(s, off, 32);
            float inv = 1.f / s;
            row[lane] = e0 * inv; row[lane + 32] = e1 * inv;
        }
        __syncthreads();

        // ---------------- Phase B: per-batch delta-rule on LDS Wf ----------
        {
            const int b   = wv >> 1;
            const int row = (wv & 1) * 32 + lane;
            const float* kr = sk + b * 64;
            const float* qr = sq + b * 64;
            float* W = sW + b * 4160 + row * 65;      // stride 65: conflict-free
            float vold = 0.f;
#pragma unroll 8
            for (int e = 0; e < 64; ++e) vold += W[e] * kr[e];
            const float delta = sbeta[b] * (sv[b * 64 + row] - vold);
            float hacc = 0.f;
#pragma unroll 8
            for (int e = 0; e < 64; ++e) {
                float w = W[e] + delta * kr[e];
                W[e] = w;
                hacc += w * qr[e];
            }
            const float hn = tanhf(hacc);
            sh[b * 66 + row] = hn;
            outs[((size_t)(t * BSZ + b0 + b)) * DIM + h * DH + row] = (_Float16)hn;
        }
        __syncthreads();
    }
}

// =====================================================================
// GEMM2: out[8192][1024] = x + outs_f16[8192][1024] @ Wout[1024][1024]
// Bt = W_out^T; exact tiling (64 tiles), branch-free.
// =====================================================================
__global__ __launch_bounds__(128) void gemm_out_kernel(const _Float16* __restrict__ A,
                                                       const _Float16* __restrict__ Bt,
                                                       const float* __restrict__ X,
                                                       float* __restrict__ out) {
    const int lane = threadIdx.x & 31, wave = threadIdx.x >> 5;
    const int lm = lane & 15, hi = lane >> 4, koff = hi * 8;
    const int m0  = blockIdx.y * 16;
    const int nt0 = (blockIdx.x * 4 + wave) * 8;      // gridDim.x=2 -> 64 tiles exactly
    v8f acc[8] = {};
    const _Float16* aprow = A + (size_t)(m0 + lm) * DIM + koff;
    const _Float16* bbase = Bt + (size_t)(nt0 * 16 + lm) * DIM + koff;
    for (int kk = 0; kk < DIM; kk += 32) {
        v16h a;
        const _Float16* ap = aprow + kk;
#pragma unroll
        for (int i = 0; i < 8; ++i) { a[i] = ap[i]; a[8 + i] = ap[16 + i]; }
#pragma unroll
        for (int jj = 0; jj < 8; jj += 4) {
            v16h bf[4];
#pragma unroll
            for (int j = 0; j < 4; ++j) {
                const _Float16* bp = bbase + (size_t)(jj + j) * 16 * DIM + kk;
#pragma unroll
                for (int i = 0; i < 8; ++i) { bf[j][i] = bp[i]; bf[j][8 + i] = bp[16 + i]; }
            }
#pragma unroll
            for (int j = 0; j < 4; ++j)
                acc[jj + j] = __builtin_amdgcn_wmma_f32_16x16x32_f16(
                    false, a, false, bf[j], (short)0, acc[jj + j], false, false);
        }
    }
#pragma unroll
    for (int j = 0; j < 8; ++j) {
        const int col = (nt0 + j) * 16 + lm;
#pragma unroll
        for (int r = 0; r < 8; ++r) {
            const int m = r + hi * 8;
            const size_t idx = (size_t)(m0 + m) * DIM + col;
            out[idx] = X[idx] + acc[j][r];
        }
    }
}

// =====================================================================
extern "C" void kernel_launch(void* const* d_in, const int* in_sizes, int n_in,
                              void* d_out, int out_size, void* d_ws, size_t ws_size,
                              hipStream_t stream) {
    const float* x      = (const float*)d_in[0];
    const float* state0 = (const float*)d_in[1];
    const float* W_slow = (const float*)d_in[2];
    const float* R_q    = (const float*)d_in[3];
    const float* R_k    = (const float*)d_in[4];
    const float* R_v    = (const float*)d_in[5];
    const float* r_b    = (const float*)d_in[6];
    const float* ln_g   = (const float*)d_in[7];
    const float* ln_b   = (const float*)d_in[8];
    const float* W_out  = (const float*)d_in[9];
    float* out = (float*)d_out;

    char* ws = (char*)d_ws;
    _Float16* hn      = (_Float16*)(ws + OFF_HN);
    _Float16* wslowT  = (_Float16*)(ws + OFF_WSLOWT);
    _Float16* woutT   = (_Float16*)(ws + OFF_WOUTT);
    float*    qkvb    = (float*)   (ws + OFF_QKVB);
    _Float16* outs16  = (_Float16*)(ws + OFF_OUTS);

    // one-time weight transposes (f32 -> f16); W_slow^T zero-padded to 3200 rows
    transpose_f16_kernel<<<dim3(NQP / 32, DIM / 32), 256, 0, stream>>>(W_slow, wslowT, DIM, NQ, NQP);
    transpose_f16_kernel<<<dim3(DIM / 32, DIM / 32), 256, 0, stream>>>(W_out, woutT, DIM, DIM, DIM);
    ln_kernel<<<ROWS, 256, 0, stream>>>(x, ln_g, ln_b, hn);
    gemm_qkvb_kernel<<<dim3(7, ROWS / 16), 128, 0, stream>>>(hn, wslowT, qkvb);

    const int scan_lds = SM_TOTAL * 4;   // 199,168 bytes (< 320KB WGP LDS)
    (void)hipFuncSetAttribute((const void*)scan_kernel,
                              hipFuncAttributeMaxDynamicSharedMemorySize, scan_lds);
    scan_kernel<<<32, 512, scan_lds, stream>>>(qkvb, R_q, R_k, R_v, r_b, state0, outs16);

    gemm_out_kernel<<<dim3(2, ROWS / 16), 128, 0, stream>>>(outs16, woutT, x, out);
}
